// MF_24309514895861
// MI455X (gfx1250) — matrix operand order repaired and checked by reading, not compiled, not run on hardware
//
#include <hip/hip_runtime.h>
#include <hip/hip_bf16.h>
#include <stdint.h>

#define D_   100   // model dim
#define L_   50    // seq len
#define KP_  100   // items to predict per row
#define FF_  2048  // ffn hidden
#define DP_  128   // padded model dim (multiple of 32 for bf16 WMMA K)
#define MT_  64    // batch rows per workgroup in GEMM kernels

typedef __attribute__((ext_vector_type(16))) __bf16 v16bf;
typedef __attribute__((ext_vector_type(8)))  float  v8f;

__device__ __forceinline__ v8f wmma_bf16(v16bf a, v16bf b, v8f c) {
  // D = A(16x32) x B(32x16) + C, f32 accumulate
  return __builtin_amdgcn_wmma_f32_16x16x32_bf16(false, a, false, b, (short)0, c, false, false);
}

// A-fragment: 16x32 (MxK) tile, 16-bit layout per CDNA5 ISA 7.12.2.
// Per lane the data is two contiguous 16B runs -> 2x ds_load_b128.
//   lane<16: m=lane,  K pairs {0..7} at bytes 0..15, {16..23} at bytes 32..47
//   lane>=16: same but K offset +8 (base shifted by 8 elements)
__device__ __forceinline__ v16bf frag_a(const __bf16* base, int ld) {
  const int lane = threadIdx.x & 31;
  const int m    = lane & 15;
  const int half = lane >> 4;
  const uint4* p = (const uint4*)(base + m * ld + half * 8);  // 16B aligned
  union { uint4 q[2]; v16bf v; } t;
  t.q[0] = p[0];   // K = 0..7   (+half*8)
  t.q[1] = p[2];   // K = 16..23 (+half*8)
  return t.v;
}

// B-fragment: 32x16 (KxN) tile. Source stored row-major [n][k] (W[n][k] = B[k][n]).
//   lane<16: n=lane, K={0..7,16..23}; lane>=16: K={8..15,24..31} -> 2x ds_load_b128.
__device__ __forceinline__ v16bf frag_b(const __bf16* base, int ld) {
  const int lane = threadIdx.x & 31;
  const int n    = lane & 15;
  const int half = lane >> 4;
  const uint4* p = (const uint4*)(base + n * ld + half * 8);  // 16B aligned
  union { uint4 q[2]; v16bf v; } t;
  t.q[0] = p[0];
  t.q[1] = p[2];
  return t.v;
}

// Cooperatively stage an f32 weight matrix into LDS as bf16, zero-padded to 128x128.
__device__ void stage_w(__bf16* dst, const float* W, int nr, int nc, int ldW) {
  for (int i = threadIdx.x; i < DP_ * DP_; i += blockDim.x) {
    int n = i >> 7, k = i & 127;
    float v = (n < nr && k < nc) ? W[n * ldW + k] : 0.f;
    dst[i] = (__bf16)v;
  }
}

// Copy an already-padded 64x128 bf16 activation tile global -> LDS (128b granularity).
__device__ void stage_x(__bf16* dst, const __bf16* src) {
  const uint4* s = (const uint4*)src;
  uint4* d = (uint4*)dst;
  for (int i = threadIdx.x; i < MT_ * DP_ / 8; i += blockDim.x) d[i] = s[i];
}

// Stage a bias vector into zero-padded 128-entry f32 LDS (kills divergent guards).
__device__ void stage_bias(float* dst, const float* b, int n) {
  if (threadIdx.x < DP_) dst[threadIdx.x] = (threadIdx.x < n) ? b[threadIdx.x] : 0.f;
}

// ---------------------------------------------------------------------------
// Kernel 1: x[b] = sum_l item_emb_w[item_seq[b,l]]  (+ bf16 padded copy)
// One wave32 per batch row; lanes 0..24 each own 4 columns (float4 loads).
// ---------------------------------------------------------------------------
__global__ void k_gather(const int* __restrict__ seq, const float* __restrict__ emb,
                         float* __restrict__ x, __bf16* __restrict__ xpad, int B) {
  const int wrow = blockIdx.x * (blockDim.x >> 5) + (threadIdx.x >> 5);
  const int lane = threadIdx.x & 31;
  if (wrow >= B) return;
  const int* s = seq + (size_t)wrow * L_;
  const bool act = lane < 25;
  float a0 = 0.f, a1 = 0.f, a2 = 0.f, a3 = 0.f;
  int idx = s[0];
  for (int t = 0; t < L_; ++t) {
    int nidx = (t < L_ - 1) ? s[t + 1] : idx;
    __builtin_prefetch(emb + (size_t)nidx * D_ + lane * 4, 0, 1);  // global_prefetch_b8
    if (act) {
      const float4* p = (const float4*)(emb + (size_t)idx * D_ + lane * 4);
      float4 v = *p;                                // 400B rows are 16B aligned
      a0 += v.x; a1 += v.y; a2 += v.z; a3 += v.w;
    }
    idx = nidx;
  }
  if (act) {
    float4* xo = (float4*)(x + (size_t)wrow * D_ + lane * 4);
    *xo = make_float4(a0, a1, a2, a3);
  }
  // bf16 padded copy: lanes 25..31 hold zeros -> cols 100..127 zero-filled
  union { __bf16 h[4]; uint2 u; } pk;
  pk.h[0] = (__bf16)a0; pk.h[1] = (__bf16)a1; pk.h[2] = (__bf16)a2; pk.h[3] = (__bf16)a3;
  *(uint2*)(xpad + (size_t)wrow * DP_ + lane * 4) = pk.u;
}

// ---------------------------------------------------------------------------
// Kernel 2: v = x@Wv^T + bv ; attn = v@Wo^T + bo ; x1 = LN1(x + attn)
// Only the V slice (rows 200..299) of in_proj is ever used by the reference.
// ---------------------------------------------------------------------------
__global__ void k_attn_ln1(const float* __restrict__ in_proj_w, const float* __restrict__ in_proj_b,
                           const float* __restrict__ out_proj_w, const float* __restrict__ out_proj_b,
                           const float* __restrict__ ln1_g, const float* __restrict__ ln1_b,
                           const float* __restrict__ x, const __bf16* __restrict__ xpad,
                           float* __restrict__ x1, __bf16* __restrict__ x1pad) {
  extern __shared__ char smem[];
  __bf16* wv = (__bf16*)smem;            // 128x128 bf16
  __bf16* wo = wv + DP_ * DP_;           // 128x128 bf16
  __bf16* xt = wo + DP_ * DP_;           // 64x128 bf16
  __bf16* vt = xt + MT_ * DP_;           // 64x128 bf16
  float*  st = (float*)(vt + MT_ * DP_); // 64x128 f32
  float*  bv = st + MT_ * DP_;           // 128 f32 (zero padded bias)
  float*  bo = bv + DP_;                 // 128 f32

  const int row0 = blockIdx.x * MT_;
  stage_w(wv, in_proj_w + 2 * D_ * D_, D_, D_, D_);   // V slice of in_proj
  stage_w(wo, out_proj_w, D_, D_, D_);
  stage_x(xt, xpad + (size_t)row0 * DP_);
  stage_bias(bv, in_proj_b + 2 * D_, D_);
  stage_bias(bo, out_proj_b, D_);
  __syncthreads();

  const int wid  = threadIdx.x >> 5;
  const int lane = threadIdx.x & 31;
  const int nl = lane & 15, hb = (lane >> 4) * 8;

  // GEMM1: v tile (bf16 out, bias added)
#pragma unroll
  for (int i = 0; i < 4; ++i) {
    int t = wid + 8 * i, mt = t >> 3, nt = t & 7;
    v8f acc = {};
#pragma unroll
    for (int kk = 0; kk < DP_; kk += 32)
      acc = wmma_bf16(frag_a(xt + mt * 16 * DP_ + kk, DP_),
                      frag_b(wv + nt * 16 * DP_ + kk, DP_), acc);
    int n = nt * 16 + nl;
    float bias = bv[n];
#pragma unroll
    for (int r = 0; r < 8; ++r)
      vt[(mt * 16 + hb + r) * DP_ + n] = (__bf16)(acc[r] + bias);
  }
  __syncthreads();

  // GEMM2: attn tile (f32 out, bias added)
#pragma unroll
  for (int i = 0; i < 4; ++i) {
    int t = wid + 8 * i, mt = t >> 3, nt = t & 7;
    v8f acc = {};
#pragma unroll
    for (int kk = 0; kk < DP_; kk += 32)
      acc = wmma_bf16(frag_a(vt + mt * 16 * DP_ + kk, DP_),
                      frag_b(wo + nt * 16 * DP_ + kk, DP_), acc);
    int n = nt * 16 + nl;
    float bias = bo[n];
#pragma unroll
    for (int r = 0; r < 8; ++r)
      st[(mt * 16 + hb + r) * DP_ + n] = acc[r] + bias;
  }
  __syncthreads();

  // residual + LayerNorm, one thread per row (f32 precision)
  if (threadIdx.x < MT_) {
    int row = row0 + threadIdx.x;
    float* s = st + threadIdx.x * DP_;
    float mean = 0.f;
    for (int k = 0; k < D_; ++k) { s[k] += x[(size_t)row * D_ + k]; mean += s[k]; }
    mean *= (1.f / D_);
    float var = 0.f;
    for (int k = 0; k < D_; ++k) { float d = s[k] - mean; var += d * d; }
    var *= (1.f / D_);
    float inv = rsqrtf(var + 1e-5f);
    for (int k = 0; k < D_; ++k) {
      float o = (s[k] - mean) * inv * ln1_g[k] + ln1_b[k];
      x1[(size_t)row * D_ + k] = o;
      x1pad[(size_t)row * DP_ + k] = (__bf16)o;
    }
    for (int k = D_; k < DP_; ++k) x1pad[(size_t)row * DP_ + k] = (__bf16)0.f;
  }
}

// ---------------------------------------------------------------------------
// Kernel 3: fused FFN: seq = LN2(x1 + relu(x1@W1^T + b1)@W2^T + b2)
// FF looped in 128-wide chunks; hidden tile never leaves LDS; second-GEMM
// accumulators stay in VGPRs across all 16 chunks.
// ---------------------------------------------------------------------------
__global__ void k_ffn_ln2(const float* __restrict__ ffn_w1, const float* __restrict__ ffn_b1,
                          const float* __restrict__ ffn_w2, const float* __restrict__ ffn_b2,
                          const float* __restrict__ ln2_g, const float* __restrict__ ln2_b,
                          const float* __restrict__ x1, const __bf16* __restrict__ x1pad,
                          __bf16* __restrict__ seqpad) {
  extern __shared__ char smem[];
  __bf16* xt = (__bf16*)smem;            // 64x128 bf16
  __bf16* w1 = xt + MT_ * DP_;           // 128x128 bf16
  __bf16* w2 = w1 + DP_ * DP_;           // 128x128 bf16
  __bf16* ht = w2 + DP_ * DP_;           // 64x128 bf16
  float*  st = (float*)(ht + MT_ * DP_); // 64x128 f32
  float*  b2s = st + MT_ * DP_;          // 128 f32 (zero padded ffn_b2)

  const int row0 = blockIdx.x * MT_;
  stage_x(xt, x1pad + (size_t)row0 * DP_);
  stage_bias(b2s, ffn_b2, D_);

  const int wid  = threadIdx.x >> 5;
  const int lane = threadIdx.x & 31;
  const int nl = lane & 15, hb = (lane >> 4) * 8;

  v8f acc2[4] = {};

  for (int c = 0; c < FF_ / DP_; ++c) {
    // W1 chunk: rows c*128..+127, cols 0..99 (k padded)
    stage_w(w1, ffn_w1 + (size_t)c * DP_ * D_, DP_, D_, D_);
    // W2 chunk: lds[n][k] = ffn_w2[n][c*128 + k], n padded to 128
    for (int i = threadIdx.x; i < DP_ * DP_; i += blockDim.x) {
      int n = i >> 7, k = i & 127;
      float v = (n < D_) ? ffn_w2[(size_t)n * FF_ + c * DP_ + k] : 0.f;
      w2[i] = (__bf16)v;
    }
    __syncthreads();

    // GEMM-A: h = relu(x1 @ W1c^T + b1) -> bf16 LDS
#pragma unroll
    for (int i = 0; i < 4; ++i) {
      int t = wid + 8 * i, mt = t >> 3, nt = t & 7;
      v8f acc = {};
#pragma unroll
      for (int kk = 0; kk < DP_; kk += 32)
        acc = wmma_bf16(frag_a(xt + mt * 16 * DP_ + kk, DP_),
                        frag_b(w1 + nt * 16 * DP_ + kk, DP_), acc);
      int n = nt * 16 + nl;
      float bias = ffn_b1[c * DP_ + n];   // always in range (all 2048 rows valid)
#pragma unroll
      for (int r = 0; r < 8; ++r) {
        float h = acc[r] + bias;
        ht[(mt * 16 + hb + r) * DP_ + n] = (__bf16)(h > 0.f ? h : 0.f);
      }
    }
    __syncthreads();

    // GEMM-B: acc2 += h @ W2c^T (accumulators persistent in VGPRs)
#pragma unroll
    for (int i = 0; i < 4; ++i) {
      int t = wid + 8 * i, mt = t >> 3, nt = t & 7;
#pragma unroll
      for (int kk = 0; kk < DP_; kk += 32)
        acc2[i] = wmma_bf16(frag_a(ht + mt * 16 * DP_ + kk, DP_),
                            frag_b(w2 + nt * 16 * DP_ + kk, DP_), acc2[i]);
    }
    __syncthreads();   // before next chunk overwrites w1/w2/ht
  }

  // epilogue: + b2, residual, LN2
#pragma unroll
  for (int i = 0; i < 4; ++i) {
    int t = wid + 8 * i, mt = t >> 3, nt = t & 7;
    int n = nt * 16 + nl;
    float bias = b2s[n];
#pragma unroll
    for (int r = 0; r < 8; ++r)
      st[(mt * 16 + hb + r) * DP_ + n] = acc2[i][r] + bias;
  }
  __syncthreads();
  if (threadIdx.x < MT_) {
    int row = row0 + threadIdx.x;
    float* s = st + threadIdx.x * DP_;
    float mean = 0.f;
    for (int k = 0; k < D_; ++k) { s[k] += x1[(size_t)row * D_ + k]; mean += s[k]; }
    mean *= (1.f / D_);
    float var = 0.f;
    for (int k = 0; k < D_; ++k) { float d = s[k] - mean; var += d * d; }
    var *= (1.f / D_);
    float inv = rsqrtf(var + 1e-5f);
    for (int k = 0; k < D_; ++k)
      seqpad[(size_t)row * DP_ + k] = (__bf16)((s[k] - mean) * inv * ln2_g[k] + ln2_b[k]);
    for (int k = D_; k < DP_; ++k) seqpad[(size_t)row * DP_ + k] = (__bf16)0.f;
  }
}

// ---------------------------------------------------------------------------
// Kernel 4: scores = seq @ rec_emb^T (all 100 candidates), then gather
// out[b,k] = rec_b[idx] + scores[b, idx] from LDS.
// ---------------------------------------------------------------------------
__global__ void k_scores(const float* __restrict__ rec_emb_w, const float* __restrict__ rec_b_w,
                         const int* __restrict__ topredict, const __bf16* __restrict__ seqpad,
                         float* __restrict__ out) {
  extern __shared__ char smem[];
  __bf16* rw = (__bf16*)smem;            // 128x128 bf16
  __bf16* xt = rw + DP_ * DP_;           // 64x128 bf16
  float*  sc = (float*)(xt + MT_ * DP_); // 64x128 f32

  const int row0 = blockIdx.x * MT_;
  stage_w(rw, rec_emb_w, D_, D_, D_);
  stage_x(xt, seqpad + (size_t)row0 * DP_);
  __syncthreads();

  const int wid  = threadIdx.x >> 5;
  const int lane = threadIdx.x & 31;
  const int nl = lane & 15, hb = (lane >> 4) * 8;
#pragma unroll
  for (int i = 0; i < 4; ++i) {
    int t = wid + 8 * i, mt = t >> 3, nt = t & 7;
    v8f acc = {};
#pragma unroll
    for (int kk = 0; kk < DP_; kk += 32)
      acc = wmma_bf16(frag_a(xt + mt * 16 * DP_ + kk, DP_),
                      frag_b(rw + nt * 16 * DP_ + kk, DP_), acc);
    int n = nt * 16 + nl;
#pragma unroll
    for (int r = 0; r < 8; ++r)
      sc[(mt * 16 + hb + r) * DP_ + n] = acc[r];
  }
  __syncthreads();

  for (int i = threadIdx.x; i < MT_ * KP_; i += blockDim.x) {
    int row = i / KP_, k = i - row * KP_;
    int idx = topredict[(size_t)(row0 + row) * KP_ + k];   // in [0,100)
    out[(size_t)(row0 + row) * KP_ + k] = rec_b_w[idx] + sc[row * DP_ + idx];
  }
}

// ---------------------------------------------------------------------------
extern "C" void kernel_launch(void* const* d_in, const int* in_sizes, int n_in,
                              void* d_out, int out_size, void* d_ws, size_t ws_size,
                              hipStream_t stream) {
  const int*   item_seq   = (const int*)  d_in[0];
  const int*   topred     = (const int*)  d_in[1];
  const float* item_emb   = (const float*)d_in[2];
  const float* rec_emb    = (const float*)d_in[3];
  const float* rec_b      = (const float*)d_in[4];
  const float* in_proj_w  = (const float*)d_in[5];
  const float* in_proj_b  = (const float*)d_in[6];
  const float* out_proj_w = (const float*)d_in[7];
  const float* out_proj_b = (const float*)d_in[8];
  const float* ln1_g      = (const float*)d_in[9];
  const float* ln1_b      = (const float*)d_in[10];
  const float* ln2_g      = (const float*)d_in[11];
  const float* ln2_b      = (const float*)d_in[12];
  const float* ffn_w1     = (const float*)d_in[13];
  const float* ffn_b1     = (const float*)d_in[14];
  const float* ffn_w2     = (const float*)d_in[15];
  const float* ffn_b2     = (const float*)d_in[16];
  (void)n_in; (void)out_size; (void)ws_size;

  const int B = in_sizes[0] / L_;

  // workspace carving (256B aligned): ~25.7 MB total for B=16384
  char* ws = (char*)d_ws;
  size_t o = 0;
  auto carve = [&](size_t bytes) { char* p = ws + o; o = (o + bytes + 255) & ~(size_t)255; return p; };
  float*  x     = (float*) carve((size_t)B * D_  * sizeof(float));
  __bf16* xpad  = (__bf16*)carve((size_t)B * DP_ * sizeof(__bf16));
  float*  x1    = (float*) carve((size_t)B * D_  * sizeof(float));
  __bf16* x1pad = (__bf16*)carve((size_t)B * DP_ * sizeof(__bf16));
  __bf16* seqp  = (__bf16*)carve((size_t)B * DP_ * sizeof(__bf16));

  const size_t lds2 = (size_t)(2 * DP_ * DP_ + 2 * MT_ * DP_) * 2 + (size_t)MT_ * DP_ * 4
                    + 2 * DP_ * sizeof(float);                                               // ~129 KB
  const size_t lds3 = (size_t)(2 * DP_ * DP_ + 2 * MT_ * DP_) * 2 + (size_t)MT_ * DP_ * 4
                    + DP_ * sizeof(float);                                                   // ~128.5 KB
  const size_t lds4 = (size_t)(DP_ * DP_ + MT_ * DP_) * 2 + (size_t)MT_ * DP_ * 4;           // 80 KB
  hipFuncSetAttribute((const void*)k_attn_ln1, hipFuncAttributeMaxDynamicSharedMemorySize, (int)lds2);
  hipFuncSetAttribute((const void*)k_ffn_ln2,  hipFuncAttributeMaxDynamicSharedMemorySize, (int)lds3);
  hipFuncSetAttribute((const void*)k_scores,   hipFuncAttributeMaxDynamicSharedMemorySize, (int)lds4);

  dim3 blk(256);  // 8 wave32 per block
  k_gather  <<<dim3((B + 7) / 8), blk, 0,    stream>>>(item_seq, item_emb, x, xpad, B);
  k_attn_ln1<<<dim3(B / MT_),     blk, lds2, stream>>>(in_proj_w, in_proj_b, out_proj_w, out_proj_b,
                                                       ln1_g, ln1_b, x, xpad, x1, x1pad);
  k_ffn_ln2 <<<dim3(B / MT_),     blk, lds3, stream>>>(ffn_w1, ffn_b1, ffn_w2, ffn_b2,
                                                       ln2_g, ln2_b, x1, x1pad, seqp);
  k_scores  <<<dim3(B / MT_),     blk, lds4, stream>>>(rec_emb, rec_b, topred, seqp, (float*)d_out);
}
